// Model_20091857011535
// MI455X (gfx1250) — matrix operand order repaired and checked by reading, not compiled, not run on hardware
//
#include <hip/hip_runtime.h>
#include <stdint.h>
#include <math.h>

// ---------------------------------------------------------------------------
// Model dims (from reference)
// ---------------------------------------------------------------------------
#define U0v 8192
#define R0v 8192
#define I0v 16384
#define U1v 4096
#define R1v 4096
#define I1v 8192
#define U2v 2048
#define R2v 4096
#define E0v 262144
#define E1v 131072
#define EPv 65536
#define Dv  128
#define Hh  4

typedef __attribute__((ext_vector_type(16))) __bf16   bf16x16;
typedef __attribute__((ext_vector_type(8)))  float    f32x8;
typedef __attribute__((ext_vector_type(8)))  uint32_t u32x8;

static __device__ __forceinline__ float wsum32(float v) {
#pragma unroll
  for (int m = 16; m >= 1; m >>= 1) v += __shfl_xor(v, m, 32);
  return v;
}

// 8 WMMA rank-32 updates (one per 16-col tile) sharing one A fragment.
static __device__ __forceinline__ void wmma8(f32x8 (&acc)[8], bf16x16 a,
                                             const uint32_t* __restrict__ wp) {
#pragma unroll
  for (int nt = 0; nt < 8; ++nt) {
    u32x8 braw = *(const u32x8*)(wp + (size_t)nt * 256);
    bf16x16 b  = __builtin_bit_cast(bf16x16, braw);
    acc[nt] = __builtin_amdgcn_wmma_f32_16x16x32_bf16(
        false, a, false, b, (short)0, acc[nt], false, false);
  }
}

// ---------------------------------------------------------------------------
// Weight packing: W[K,128] fp32 (row-major) -> bf16 WMMA B-fragments.
// Fragment for (kt, nt, lane) = 8 dwords, contiguous, so the GEMM does one
// 32B load per lane per B tile. Layout per ISA 7.12.2:
//   lane<16  : col = nt*16+lane,    K = kt*32 + {0..7, 16..23}  (pairs low=even K)
//   lane>=16 : col = nt*16+lane-16, K = kt*32 + {8..15, 24..31}
// ---------------------------------------------------------------------------
__global__ __launch_bounds__(256) void pack_w_kernel(
    const float* __restrict__ W, int K, uint32_t* __restrict__ out) {
  const int idx   = blockIdx.x * 256 + threadIdx.x;  // kt*256 + nt*32 + lane
  const int lane  = idx & 31;
  const int nt    = (idx >> 5) & 7;
  const int kt    = idx >> 8;
  const int col   = nt * 16 + (lane & 15);
  const int kbase = kt * 32 + ((lane >> 4) << 3);
  uint32_t* o = out + (size_t)idx * 8;
#pragma unroll
  for (int j = 0; j < 8; ++j) {
    const int k0 = kbase + 2 * j + ((j >= 4) ? 8 : 0), k1 = k0 + 1;
    float f0 = (k0 < K) ? W[(size_t)k0 * 128 + col] : 0.f;
    float f1 = (k1 < K) ? W[(size_t)k1 * 128 + col] : 0.f;
    __bf16 b0 = (__bf16)f0, b1 = (__bf16)f1;
    unsigned short u0, u1;
    __builtin_memcpy(&u0, &b0, 2);
    __builtin_memcpy(&u1, &b1, 2);
    o[j] = (uint32_t)u0 | ((uint32_t)u1 << 16);
  }
}

// ---------------------------------------------------------------------------
// Workhorse GEMM: O[M,128] = post( X[M,K] @ W + bias )
//   post: optional ReLU, optional addend (per-row [M,128] or broadcast [128]),
//         optional row L2-normalize.
// One wave owns a 16-row x 128-col strip: 8 v8f accumulators,
// K-loop in steps of 32 with v_wmma_f32_16x16x32_bf16.
// A fragment (ISA 7.12.2, 16-bit A 16x32) per lane: 16 bf16,
//   elems [0..7] = K kb..kb+7, elems [8..15] = K kb+16..kb+23,
//   kb = kt*32 + 8*(lane>=16).
// ---------------------------------------------------------------------------
__global__ __launch_bounds__(128, 1) void gemm128_kernel(
    const float* __restrict__ X, int K, int M,
    const uint32_t* __restrict__ Wpk, int Kt,
    const float* __restrict__ bias,
    const float* __restrict__ addend, int addend_bcast,
    float* __restrict__ O, int do_relu, int do_l2n) {
  const int lane    = threadIdx.x & 31;
  const int rowBase = (blockIdx.x * 4 + (threadIdx.x >> 5)) * 16;
  if (rowBase >= M) return;
  const int hl  = lane >> 4;  // half-wave: 0 or 1
  const int l15 = lane & 15;
  const float* __restrict__ Xrow = X + (size_t)(rowBase + l15) * K;

  f32x8 acc[8];
#pragma unroll
  for (int nt = 0; nt < 8; ++nt)
#pragma unroll
    for (int r = 0; r < 8; ++r) acc[nt][r] = 0.f;

  const int ktFull = K >> 5;  // tiles with no K-boundary interaction

  if ((K & 3) == 0) {
    // Vector fast path: 4x global_load_b128 per lane per tile.
    for (int kt = 0; kt < ktFull; ++kt) {
      const int kb = (kt << 5) + hl * 8;
      __builtin_prefetch(Xrow + kb + 128, 0, 3);
      const float4* p = (const float4*)(Xrow + kb);
      const float4 u0 = p[0], u1 = p[1];
      const float4* q = (const float4*)(Xrow + kb + 16);
      const float4 w0 = q[0], w1 = q[1];
      bf16x16 a;
      a[0]  = (__bf16)u0.x; a[1]  = (__bf16)u0.y;
      a[2]  = (__bf16)u0.z; a[3]  = (__bf16)u0.w;
      a[4]  = (__bf16)u1.x; a[5]  = (__bf16)u1.y;
      a[6]  = (__bf16)u1.z; a[7]  = (__bf16)u1.w;
      a[8]  = (__bf16)w0.x; a[9]  = (__bf16)w0.y;
      a[10] = (__bf16)w0.z; a[11] = (__bf16)w0.w;
      a[12] = (__bf16)w1.x; a[13] = (__bf16)w1.y;
      a[14] = (__bf16)w1.z; a[15] = (__bf16)w1.w;
      wmma8(acc, a, Wpk + (((size_t)kt << 8) + lane) * 8);
    }
  } else {
    // Scalar (but unguarded, branch-free) path for K not 0 mod 4.
    for (int kt = 0; kt < ktFull; ++kt) {
      const int kb = (kt << 5) + hl * 8;
      bf16x16 a;
#pragma unroll
      for (int j = 0; j < 8; ++j) {
        a[j]     = (__bf16)Xrow[kb + j];
        a[8 + j] = (__bf16)Xrow[kb + 16 + j];
      }
      wmma8(acc, a, Wpk + (((size_t)kt << 8) + lane) * 8);
    }
  }

  if (ktFull < Kt) {
    // Tail tile: branch-free guard (clamped index + cndmask zeroing).
    const int kb = (ktFull << 5) + hl * 8;
    bf16x16 a;
#pragma unroll
    for (int j = 0; j < 8; ++j) {
      const int k0 = kb + j, k1 = kb + 16 + j;
      const int c0 = min(k0, K - 1), c1 = min(k1, K - 1);
      a[j]     = (__bf16)(Xrow[c0] * ((k0 < K) ? 1.f : 0.f));
      a[8 + j] = (__bf16)(Xrow[c1] * ((k1 < K) ? 1.f : 0.f));
    }
    wmma8(acc, a, Wpk + (((size_t)ktFull << 8) + lane) * 8);
  }

  // Epilogue. C layout: acc[nt][r] -> row rowBase+r+8*hl, col nt*16+l15.
  float ss[8];
#pragma unroll
  for (int r = 0; r < 8; ++r) ss[r] = 0.f;
#pragma unroll
  for (int nt = 0; nt < 8; ++nt) {
    const int col = nt * 16 + l15;
#pragma unroll
    for (int r = 0; r < 8; ++r) {
      float v = acc[nt][r];
      if (bias) v += bias[col];
      if (do_relu) v = fmaxf(v, 0.f);
      if (addend) {
        const int row = rowBase + r + hl * 8;
        v += addend_bcast ? addend[col] : addend[(size_t)row * 128 + col];
      }
      acc[nt][r] = v;
      ss[r] += v * v;
    }
  }
  float scl[8];
#pragma unroll
  for (int r = 0; r < 8; ++r) {
    float s = ss[r];
#pragma unroll
    for (int m = 8; m >= 1; m >>= 1) s += __shfl_xor(s, m, 32);  // 16-lane half
    scl[r] = do_l2n ? (1.f / fmaxf(sqrtf(s), 1e-12f)) : 1.f;
  }
#pragma unroll
  for (int nt = 0; nt < 8; ++nt) {
    const int col = nt * 16 + l15;
#pragma unroll
    for (int r = 0; r < 8; ++r) {
      const int row = rowBase + r + hl * 8;
      O[(size_t)row * 128 + col] = acc[nt][r] * scl[r];
    }
  }
}

// ---------------------------------------------------------------------------
// Edge scatter-add (segment mean numerator + counts). One wave per edge,
// lane moves 4 floats; global_atomic_add_f32.
// ---------------------------------------------------------------------------
__global__ __launch_bounds__(256) void scatter_add_kernel(
    const float* __restrict__ src, const int* __restrict__ si,
    const int* __restrict__ di, int E, float* __restrict__ sums,
    float* __restrict__ cnt) {
  const int g = blockIdx.x * 256 + threadIdx.x;
  const int e = g >> 5, lane = g & 31;
  if (e >= E) return;
  const int s = si[e], d = di[e];
  const float* sp = src + (size_t)s * 128 + lane * 4;
  float* dp = sums + (size_t)d * 128 + lane * 4;
  atomicAdd(dp + 0, sp[0]);
  atomicAdd(dp + 1, sp[1]);
  atomicAdd(dp + 2, sp[2]);
  atomicAdd(dp + 3, sp[3]);
  if (lane == 0) atomicAdd(cnt + d, 1.0f);
}

// out = relu(self + sumA/max(cntA,1) [+ sumB/max(cntB,1)])
__global__ __launch_bounds__(256) void seg_combine_relu_kernel(
    const float* __restrict__ self, const float* __restrict__ sumA,
    const float* __restrict__ cntA, const float* __restrict__ sumB,
    const float* __restrict__ cntB, float* __restrict__ out, int n) {
  const int i = blockIdx.x * 256 + threadIdx.x;
  if (i >= n) return;
  const int row = i >> 7;
  float a = self[i] + sumA[i] / fmaxf(cntA[row], 1.f);
  if (sumB) a += sumB[i] / fmaxf(cntB[row], 1.f);
  out[i] = fmaxf(a, 0.f);
}

__global__ __launch_bounds__(256) void zero_kernel(float* __restrict__ p, int n) {
  const int i = blockIdx.x * 256 + threadIdx.x;
  if (i < n) p[i] = 0.f;
}
__global__ __launch_bounds__(256) void copy_kernel(float* __restrict__ d,
                                                   const float* __restrict__ s,
                                                   int n) {
  const int i = blockIdx.x * 256 + threadIdx.x;
  if (i < n) d[i] = s[i];
}
__global__ __launch_bounds__(256) void add2_kernel(float* __restrict__ d,
                                                   const float* __restrict__ a,
                                                   const float* __restrict__ b,
                                                   int n) {
  const int i = blockIdx.x * 256 + threadIdx.x;
  if (i < n) d[i] = a[i] + b[i];
}

// S[w,:] = l2n(ingre_dst[nbr_idx[w]] * mask[w]); one wave per row.
__global__ __launch_bounds__(256) void build_S_kernel(
    const float* __restrict__ G, const int* __restrict__ idx,
    const int* __restrict__ msk, float* __restrict__ S, int n) {
  const int g = blockIdx.x * 256 + threadIdx.x;
  const int w = g >> 5, lane = g & 31;
  if (w >= n) return;
  const float m  = (float)msk[w];
  const float* p = G + (size_t)idx[w] * 128 + lane * 4;
  float v0 = p[0] * m, v1 = p[1] * m, v2 = p[2] * m, v3 = p[3] * m;
  const float ss  = wsum32(v0 * v0 + v1 * v1 + v2 * v2 + v3 * v3);
  const float scl = 1.f / fmaxf(sqrtf(ss), 1e-12f);
  float* o = S + (size_t)w * 128 + lane * 4;
  o[0] = v0 * scl; o[1] = v1 * scl; o[2] = v2 * scl; o[3] = v3 * scl;
}

// Self-attention over 5 tokens, per (recipe, head); lane = head-dim.
__global__ __launch_bounds__(128) void attn_self5_kernel(
    const float* __restrict__ Q, const float* __restrict__ Kk,
    const float* __restrict__ V, float* __restrict__ O) {
  const int w = (blockIdx.x * blockDim.x + threadIdx.x) >> 5;
  const int lane = threadIdx.x & 31;
  const int r = w >> 2, h = w & 3;
  if (r >= R2v) return;
  const size_t base = (size_t)r * 5 * 128 + h * 32 + lane;
  float q[5], k[5], v[5];
#pragma unroll
  for (int j = 0; j < 5; ++j) {
    q[j] = Q[base + j * 128];
    k[j] = Kk[base + j * 128];
    v[j] = V[base + j * 128];
  }
  const float scale = 0.17677669529663687f;  // 1/sqrt(32)
#pragma unroll
  for (int i = 0; i < 5; ++i) {
    float s[5];
#pragma unroll
    for (int j = 0; j < 5; ++j) s[j] = wsum32(q[i] * k[j]) * scale;
    float m = s[0];
#pragma unroll
    for (int j = 1; j < 5; ++j) m = fmaxf(m, s[j]);
    float e[5], sum = 0.f;
#pragma unroll
    for (int j = 0; j < 5; ++j) { e[j] = __expf(s[j] - m); sum += e[j]; }
    float oi = 0.f;
#pragma unroll
    for (int j = 0; j < 5; ++j) oi += (e[j] / sum) * v[j];
    O[base + i * 128] = oi;
  }
}

// Pooling attention: shared query q2[128] vs 5 keys per (recipe, head).
__global__ __launch_bounds__(128) void attn_pool_kernel(
    const float* __restrict__ q2, const float* __restrict__ K2,
    const float* __restrict__ V2, float* __restrict__ O2) {
  const int w = (blockIdx.x * blockDim.x + threadIdx.x) >> 5;
  const int lane = threadIdx.x & 31;
  const int r = w >> 2, h = w & 3;
  if (r >= R2v) return;
  const size_t base = (size_t)r * 5 * 128 + h * 32 + lane;
  const float qv = q2[h * 32 + lane];
  const float scale = 0.17677669529663687f;
  float s[5];
#pragma unroll
  for (int j = 0; j < 5; ++j) s[j] = wsum32(qv * K2[base + j * 128]) * scale;
  float m = s[0];
#pragma unroll
  for (int j = 1; j < 5; ++j) m = fmaxf(m, s[j]);
  float e[5], sum = 0.f;
#pragma unroll
  for (int j = 0; j < 5; ++j) { e[j] = __expf(s[j] - m); sum += e[j]; }
  float o = 0.f;
#pragma unroll
  for (int j = 0; j < 5; ++j) o += (e[j] / sum) * V2[base + j * 128];
  O2[(size_t)r * 128 + h * 32 + lane] = o;
}

// y[1,128] = x[1,128] @ W[128,128]  (tiny, one block)
__global__ __launch_bounds__(128) void vecmat128_kernel(
    const float* __restrict__ x, const float* __restrict__ W,
    float* __restrict__ y) {
  const int n = threadIdx.x;
  float s = 0.f;
  for (int k = 0; k < 128; ++k) s += x[k] * W[(size_t)k * 128 + n];
  y[n] = s;
}

// out[e] = dot(A[si[e]], B[di[e]]); one wave per pair.
__global__ __launch_bounds__(256) void pair_dot_kernel(
    const float* __restrict__ A, const float* __restrict__ B,
    const int* __restrict__ si, const int* __restrict__ di, int E,
    float* __restrict__ out) {
  const int g = blockIdx.x * 256 + threadIdx.x;
  const int e = g >> 5, lane = g & 31;
  if (e >= E) return;
  const float* a = A + (size_t)si[e] * 128 + lane * 4;
  const float* b = B + (size_t)di[e] * 128 + lane * 4;
  float p = a[0] * b[0] + a[1] * b[1] + a[2] * b[2] + a[3] * b[3];
  p = wsum32(p);
  if (lane == 0) out[e] = p;
}

// ---------------------------------------------------------------------------
// Orchestration
// ---------------------------------------------------------------------------
extern "C" void kernel_launch(void* const* d_in, const int* in_sizes, int n_in,
                              void* d_out, int out_size, void* d_ws,
                              size_t ws_size, hipStream_t stream) {
  (void)in_sizes; (void)n_in; (void)out_size; (void)ws_size;

  const float* user      = (const float*)d_in[0];
  const float* instr     = (const float*)d_in[1];
  const float* ingredient= (const float*)d_in[2];
  const float* ingre_dst = (const float*)d_in[3];
  const float* Wu = (const float*)d_in[4];  const float* bu = (const float*)d_in[5];
  const float* Wi = (const float*)d_in[6];  const float* bi = (const float*)d_in[7];
  const float* Wg = (const float*)d_in[8];  const float* bg = (const float*)d_in[9];
  const float* Wc = (const float*)d_in[10]; const float* bc = (const float*)d_in[11];
  const float* gw0 = (const float*)d_in[12];
  const float* gw1 = (const float*)d_in[13];
  const float* stw = (const float*)d_in[14];
  const float* st_seed = (const float*)d_in[15];
  const int* b0_ur_s = (const int*)d_in[16]; const int* b0_ur_d = (const int*)d_in[17];
  const int* b0_ir_s = (const int*)d_in[18]; const int* b0_ir_d = (const int*)d_in[19];
  const int* b0_ru_s = (const int*)d_in[20]; const int* b0_ru_d = (const int*)d_in[21];
  const int* b0_ri_s = (const int*)d_in[22]; const int* b0_ri_d = (const int*)d_in[23];
  const int* b1_ur_s = (const int*)d_in[24]; const int* b1_ur_d = (const int*)d_in[25];
  const int* b1_ir_s = (const int*)d_in[26]; const int* b1_ir_d = (const int*)d_in[27];
  const int* b1_ru_s = (const int*)d_in[28]; const int* b1_ru_d = (const int*)d_in[29];
  const int* pos_s = (const int*)d_in[30]; const int* pos_d = (const int*)d_in[31];
  const int* neg_s = (const int*)d_in[32]; const int* neg_d = (const int*)d_in[33];
  const int* nbr_idx  = (const int*)d_in[34];
  const int* nbr_mask = (const int*)d_in[35];

  float* out_pos = (float*)d_out;
  float* out_neg = out_pos + EPv;
  float* out_x1u = out_neg + EPv;
  float* out_x1r = out_x1u + U2v * Dv;
  float* out_x2u = out_x1r + R2v * Dv;
  float* out_x2r = out_x2u + U2v * Dv;

  char* wsb = (char*)d_ws;
  size_t off = 0;
  auto alloc = [&](size_t bytes) -> void* {
    void* p = wsb + off;
    off += (bytes + 255) & ~(size_t)255;
    return p;
  };
  auto fb = [&](size_t n) -> float* { return (float*)alloc(n * 4); };

  auto pack = [&](const float* W, int K) -> const uint32_t* {
    const int Kt = (K + 31) / 32;
    uint32_t* p = (uint32_t*)alloc((size_t)Kt * 8192);
    pack_w_kernel<<<Kt, 256, 0, stream>>>(W, K, p);
    return p;
  };
  auto gemm = [&](const float* X, int K, int M, const uint32_t* Wpk,
                  const float* bias, const float* addend, int bcast, float* O,
                  int relu, int l2n) {
    gemm128_kernel<<<M / 64, 128, 0, stream>>>(X, K, M, Wpk, (K + 31) / 32,
                                               bias, addend, bcast, O, relu, l2n);
  };
  auto zero = [&](float* p, int n) {
    zero_kernel<<<(n + 255) / 256, 256, 0, stream>>>(p, n);
  };
  auto scat = [&](const float* src, const int* si, const int* di, int E,
                  float* sums, float* cnt) {
    scatter_add_kernel<<<E / 8, 256, 0, stream>>>(src, si, di, E, sums, cnt);
  };

  // --- pack all weights into WMMA B-fragment form (bf16) ---
  const uint32_t* pWu = pack(Wu, 300);
  const uint32_t* pWi = pack(Wi, 1024);
  const uint32_t* pWg = pack(Wg, 46);
  const uint32_t* pWc = pack(Wc, 128);
  const uint32_t* pW0[7]; for (int j = 0; j < 7; ++j) pW0[j] = pack(gw0 + (size_t)j * 16384, 128);
  const uint32_t* pW1[5]; for (int j = 0; j < 5; ++j) pW1[j] = pack(gw1 + (size_t)j * 16384, 128);
  const uint32_t* pST[10]; for (int j = 0; j < 10; ++j) pST[j] = pack(stw + (size_t)j * 16384, 128);

  // --- buffers ---
  float* hu  = fb((size_t)U0v * Dv);
  float* hr  = fb((size_t)R0v * Dv);
  float* hin = fb((size_t)I0v * Dv);
  float* pu3 = fb((size_t)U0v * Dv);
  float* pi4 = fb((size_t)I0v * Dv);
  float* pr5 = fb((size_t)R0v * Dv);
  float* pr6 = fb((size_t)R0v * Dv);
  float* selfu = fb((size_t)U1v * Dv);
  float* selfr = fb((size_t)R1v * Dv);
  float* selfi = fb((size_t)I1v * Dv);
  float* sumsA = fb((size_t)R1v * Dv); float* cntA = fb(I1v);
  float* sumsB = fb((size_t)R1v * Dv); float* cntB = fb(I1v);
  float* sumsC = fb((size_t)U1v * Dv); float* cntC = fb(I1v);
  float* sumsD = fb((size_t)I1v * Dv); float* cntD = fb(I1v);
  float* hu1 = fb((size_t)U1v * Dv);
  float* hr1 = fb((size_t)R1v * Dv);
  float* hi1 = fb((size_t)I1v * Dv);
  float* pu_1 = fb((size_t)U1v * Dv);
  float* pi_1 = fb((size_t)I1v * Dv);
  float* pr_1 = fb((size_t)R1v * Dv);
  float* selfu2 = fb((size_t)U2v * Dv);
  float* selfr2 = fb((size_t)R2v * Dv);
  float* xu = fb((size_t)U2v * Dv);
  float* xr = fb((size_t)R2v * Dv);
  const int MS = R2v * 5;  // 20480 set-transformer rows
  float* S  = fb((size_t)MS * Dv);
  float* q1 = fb((size_t)MS * Dv);
  float* k1 = fb((size_t)MS * Dv);
  float* v1 = fb((size_t)MS * Dv);
  float* o1 = fb((size_t)MS * Dv);
  float* h1 = fb((size_t)MS * Dv);
  float* Xs = fb((size_t)MS * Dv);
  float* k2 = q1;  // reuse: q1 dead once o1 is written, k2 written after
  float* v2 = k1;  // reuse
  float* q2 = fb(Dv);
  float* o2 = fb((size_t)R2v * Dv);
  float* h2 = fb((size_t)R2v * Dv);
  float* pooled = fb((size_t)R2v * Dv);
  float* tmp = fb((size_t)R2v * Dv);
  float* xrf = fb((size_t)R2v * Dv);

  // --- feature projections: l2n(relu(X @ W + b)) ---
  gemm(user,       300,  U0v, pWu, bu, nullptr, 0, hu,  1, 1);
  gemm(instr,      1024, R0v, pWi, bi, nullptr, 0, hr,  1, 1);
  gemm(ingredient, 46,   I0v, pWg, bg, nullptr, 0, hin, 1, 1);

  // --- GNN block 0: hoist per-edge matmuls to per-node GEMMs ---
  gemm(hu,  128, U0v, pW0[3], nullptr, nullptr, 0, pu3, 0, 0);
  gemm(hin, 128, I0v, pW0[4], nullptr, nullptr, 0, pi4, 0, 0);
  gemm(hr,  128, R0v, pW0[5], nullptr, nullptr, 0, pr5, 0, 0);
  gemm(hr,  128, R0v, pW0[6], nullptr, nullptr, 0, pr6, 0, 0);
  gemm(hu,  128, U1v, pW0[0], nullptr, nullptr, 0, selfu, 0, 0);
  gemm(hr,  128, R1v, pW0[1], nullptr, nullptr, 0, selfr, 0, 0);
  gemm(hin, 128, I1v, pW0[2], nullptr, nullptr, 0, selfi, 0, 0);

  zero(sumsA, R1v * Dv); zero(cntA, R1v);
  zero(sumsB, R1v * Dv); zero(cntB, R1v);
  scat(pu3, b0_ur_s, b0_ur_d, E0v, sumsA, cntA);
  scat(pi4, b0_ir_s, b0_ir_d, E0v, sumsB, cntB);
  seg_combine_relu_kernel<<<(R1v * Dv + 255) / 256, 256, 0, stream>>>(
      selfr, sumsA, cntA, sumsB, cntB, hr1, R1v * Dv);
  zero(sumsC, U1v * Dv); zero(cntC, U1v);
  scat(pr5, b0_ru_s, b0_ru_d, E0v, sumsC, cntC);
  seg_combine_relu_kernel<<<(U1v * Dv + 255) / 256, 256, 0, stream>>>(
      selfu, sumsC, cntC, nullptr, nullptr, hu1, U1v * Dv);
  zero(sumsD, I1v * Dv); zero(cntD, I1v);
  scat(pr6, b0_ri_s, b0_ri_d, E0v, sumsD, cntD);
  seg_combine_relu_kernel<<<(I1v * Dv + 255) / 256, 256, 0, stream>>>(
      selfi, sumsD, cntD, nullptr, nullptr, hi1, I1v * Dv);

  // --- GNN block 1 ---
  gemm(hu1, 128, U1v, pW1[2], nullptr, nullptr, 0, pu_1, 0, 0);
  gemm(hi1, 128, I1v, pW1[3], nullptr, nullptr, 0, pi_1, 0, 0);
  gemm(hr1, 128, R1v, pW1[4], nullptr, nullptr, 0, pr_1, 0, 0);
  gemm(hu1, 128, U2v, pW1[0], nullptr, nullptr, 0, selfu2, 0, 0);
  gemm(hr1, 128, R2v, pW1[1], nullptr, nullptr, 0, selfr2, 0, 0);

  zero(sumsA, R2v * Dv); zero(cntA, R2v);
  zero(sumsB, R2v * Dv); zero(cntB, R2v);
  scat(pu_1, b1_ur_s, b1_ur_d, E1v, sumsA, cntA);
  scat(pi_1, b1_ir_s, b1_ir_d, E1v, sumsB, cntB);
  seg_combine_relu_kernel<<<(R2v * Dv + 255) / 256, 256, 0, stream>>>(
      selfr2, sumsA, cntA, sumsB, cntB, xr, R2v * Dv);
  zero(sumsC, U2v * Dv); zero(cntC, U2v);
  scat(pr_1, b1_ru_s, b1_ru_d, E1v, sumsC, cntC);
  seg_combine_relu_kernel<<<(U2v * Dv + 255) / 256, 256, 0, stream>>>(
      selfu2, sumsC, cntC, nullptr, nullptr, xu, U2v * Dv);

  // --- set transformer over 5 neighbors per recipe ---
  build_S_kernel<<<(MS * 32) / 256, 256, 0, stream>>>(ingre_dst, nbr_idx,
                                                      nbr_mask, S, MS);
  gemm(S, 128, MS, pST[0], nullptr, nullptr, 0, q1, 0, 0);
  gemm(S, 128, MS, pST[1], nullptr, nullptr, 0, k1, 0, 0);
  gemm(S, 128, MS, pST[2], nullptr, nullptr, 0, v1, 0, 0);
  attn_self5_kernel<<<(R2v * Hh) / 4, 128, 0, stream>>>(q1, k1, v1, o1);
  gemm(o1, 128, MS, pST[3], nullptr, S,  0, h1, 0, 0);   // h = S + o@wo
  gemm(h1, 128, MS, pST[4], nullptr, h1, 0, Xs, 1, 0);   // Xs = h + relu(h@w4)
  gemm(Xs, 128, MS, pST[6], nullptr, nullptr, 0, k2, 0, 0);
  gemm(Xs, 128, MS, pST[7], nullptr, nullptr, 0, v2, 0, 0);
  vecmat128_kernel<<<1, 128, 0, stream>>>(st_seed, stw + 5 * 16384, q2);
  attn_pool_kernel<<<(R2v * Hh) / 4, 128, 0, stream>>>(q2, k2, v2, o2);
  gemm(o2, 128, R2v, pST[8], nullptr, st_seed, 1, h2, 0, 0);  // seed + o@wo
  gemm(h2, 128, R2v, pST[9], nullptr, h2, 0, pooled, 1, 1);   // l2n(h+relu(h@w9))

  // --- fuse: xr_f = relu((pooled + xr) @ Wc + bc) ---
  add2_kernel<<<(R2v * Dv + 255) / 256, 256, 0, stream>>>(tmp, pooled, xr,
                                                          R2v * Dv);
  gemm(tmp, 128, R2v, pWc, bc, nullptr, 0, xrf, 1, 0);

  // --- scoring + outputs ---
  pair_dot_kernel<<<(EPv * 32) / 256, 256, 0, stream>>>(xu, xrf, pos_s, pos_d,
                                                        EPv, out_pos);
  pair_dot_kernel<<<(EPv * 32) / 256, 256, 0, stream>>>(xu, xrf, neg_s, neg_d,
                                                        EPv, out_neg);
  copy_kernel<<<(U2v * Dv + 255) / 256, 256, 0, stream>>>(out_x1u, xu, U2v * Dv);
  copy_kernel<<<(R2v * Dv + 255) / 256, 256, 0, stream>>>(out_x1r, xr, R2v * Dv);
  copy_kernel<<<(U2v * Dv + 255) / 256, 256, 0, stream>>>(out_x2u, xu, U2v * Dv);
  copy_kernel<<<(R2v * Dv + 255) / 256, 256, 0, stream>>>(out_x2r, xr, R2v * Dv);
}
